// MultiHeadCrossAttention_33552284516909
// MI455X (gfx1250) — matrix operand order repaired and checked by reading, not compiled, not run on hardware
//
#include <hip/hip_runtime.h>

#define D_MODEL 1024
#define NHEADS  16
#define HDIM    64
#define BATCH   2
#define SEQ     2048

typedef __attribute__((ext_vector_type(8)))  __bf16 v8bf;
typedef __attribute__((ext_vector_type(16))) __bf16 v16bf;
typedef __attribute__((ext_vector_type(8)))  float  v8f;

static __device__ __forceinline__ v8f zero8() {
    v8f z;
#pragma unroll
    for (int i = 0; i < 8; ++i) z[i] = 0.0f;
    return z;
}

// Build a 16-element bf16 fragment from two 16B-aligned 8-element chunks.
// A-fragment: off2=8  -> one contiguous 16-element run.
// B-fragment: off2=8  -> one contiguous 16-element run.
static __device__ __forceinline__ v16bf ld_frag(const __bf16* p, int off2) {
    v8bf lo = *(const v8bf*)p;
    v8bf hi = *(const v8bf*)(p + off2);
    v16bf r;
#pragma unroll
    for (int i = 0; i < 8; ++i) { r[i] = lo[i]; r[8 + i] = hi[i]; }
    return r;
}

static __device__ __forceinline__ v8f wmma_bf16(v16bf a, v16bf b, v8f c) {
    return __builtin_amdgcn_wmma_f32_16x16x32_bf16(false, a, false, b, (short)0, c,
                                                   false, false);
}

// ---------------------------------------------------------------------------
// WMMA GEMM:  C[M,N] = A[M,K] @ W[K,N] + bias     (compile-time specialized)
// MODE 0: store bf16 row-major (outb, ld=N)
// MODE 1: kv scatter -> K[b,h,s,d] (Kout) and V^T[b,h,d,s] (Vtout)
// MODE 2: store f32 row-major (outf, ld=N)
// ABF: A operand is bf16 (else f32, converted while staging).
// Block: 256 threads = 8 waves (4 M-strips x 2 N-strips of 32x32).
// Block tile 128x64, K staged 32 at a time, software-pipelined global->LDS.
// ---------------------------------------------------------------------------
template <int MODE, bool ABF>
__global__ __launch_bounds__(256)
void gemm_wmma_kernel(const float* __restrict__ Af, const __bf16* __restrict__ Ab,
                      const float* __restrict__ W, const float* __restrict__ bias,
                      __bf16* __restrict__ outb, float* __restrict__ outf,
                      __bf16* __restrict__ Kout, __bf16* __restrict__ Vtout,
                      int M, int N, int K)
{
    __shared__ __align__(16) __bf16 As[128 * 40];   // row-major [128][32], pad 40
    __shared__ __align__(16) __bf16 Bs[64 * 40];    // transposed [n][k], pad 40

    const int tid  = threadIdx.x;
    const int m0   = blockIdx.x * 128;
    const int n0   = blockIdx.y * 64;
    const int lane = tid & 31;
    const int wv   = tid >> 5;
    const int wm   = wv & 3;       // 0..3 : M strip of 32
    const int wn   = wv >> 2;      // 0..1 : N strip of 32
    const int half = lane >> 4;    // 0/1
    const int lm   = lane & 15;

    // per-thread staging coordinates
    const int arow = tid >> 1;          // A: 128 rows x 32 cols, 16 elems/thread
    const int ahc  = (tid & 1) * 16;
    const int bkk  = tid >> 3;          // B: 32 k-rows x 64 n-cols, 8 elems/thread
    const int bnc  = (tid & 7) * 8;

    // staging registers (next tile in flight while current tile computes)
    v8bf   aRb0, aRb1;                  // ABF path
    float4 aRf[4];                      // f32 path
    float4 bRf[2];

    auto load_tile = [&](int k0) {
        if constexpr (ABF) {
            const __bf16* s = Ab + (size_t)(m0 + arow) * K + k0 + ahc;
            aRb0 = *(const v8bf*)s;
            aRb1 = *(const v8bf*)(s + 8);
        } else {
            const float* s = Af + (size_t)(m0 + arow) * K + k0 + ahc;
#pragma unroll
            for (int u = 0; u < 4; ++u) aRf[u] = *(const float4*)(s + u * 4);
        }
        const float* ws = W + (size_t)(k0 + bkk) * N + n0 + bnc;
        bRf[0] = *(const float4*)ws;
        bRf[1] = *(const float4*)(ws + 4);
    };

    auto store_tile = [&]() {
        __bf16* dst = &As[arow * 40 + ahc];
        if constexpr (ABF) {
            *(v8bf*)dst       = aRb0;
            *(v8bf*)(dst + 8) = aRb1;
        } else {
#pragma unroll
            for (int u = 0; u < 4; ++u) {
                dst[u * 4 + 0] = (__bf16)aRf[u].x;
                dst[u * 4 + 1] = (__bf16)aRf[u].y;
                dst[u * 4 + 2] = (__bf16)aRf[u].z;
                dst[u * 4 + 3] = (__bf16)aRf[u].w;
            }
        }
#pragma unroll
        for (int u = 0; u < 2; ++u) {
            Bs[(bnc + u * 4 + 0) * 40 + bkk] = (__bf16)bRf[u].x;
            Bs[(bnc + u * 4 + 1) * 40 + bkk] = (__bf16)bRf[u].y;
            Bs[(bnc + u * 4 + 2) * 40 + bkk] = (__bf16)bRf[u].z;
            Bs[(bnc + u * 4 + 3) * 40 + bkk] = (__bf16)bRf[u].w;
        }
    };

    v8f acc[2][2];
#pragma unroll
    for (int i = 0; i < 2; ++i)
#pragma unroll
        for (int j = 0; j < 2; ++j) acc[i][j] = zero8();

    load_tile(0);                       // prologue: tile 0 into registers

    for (int k0 = 0; k0 < K; k0 += 32) {
        store_tile();                   // regs -> LDS (tile k0)
        __syncthreads();

        if (k0 + 32 < K) load_tile(k0 + 32);   // next tile in flight

        v16bf a[2], bfr[2];
#pragma unroll
        for (int i = 0; i < 2; ++i)
            a[i] = ld_frag(&As[(wm * 32 + i * 16 + lm) * 40 + half * 16], 8);
#pragma unroll
        for (int j = 0; j < 2; ++j)
            bfr[j] = ld_frag(&Bs[(wn * 32 + j * 16 + lm) * 40 + half * 16], 8);
#pragma unroll
        for (int i = 0; i < 2; ++i)
#pragma unroll
            for (int j = 0; j < 2; ++j)
                acc[i][j] = wmma_bf16(a[i], bfr[j], acc[i][j]);

        __syncthreads();                // LDS consumed; safe to overwrite
    }

    // ---- epilogue: element e of acc -> row (i*16 + half*8 + e), col (j*16 + lm)
#pragma unroll
    for (int i = 0; i < 2; ++i) {
        const int mbase = m0 + wm * 32 + i * 16 + half * 8;
#pragma unroll
        for (int j = 0; j < 2; ++j) {
            const int ncol = n0 + wn * 32 + j * 16 + lm;
            const float bv = bias[ncol];
#pragma unroll
            for (int e = 0; e < 8; ++e) {
                const float v = acc[i][j][e] + bv;
                const int mrow = mbase + e;
                if constexpr (MODE == 0) {
                    outb[(size_t)mrow * N + ncol] = (__bf16)v;
                } else if constexpr (MODE == 2) {
                    outf[(size_t)mrow * N + ncol] = v;
                } else { // kv scatter
                    const int hh = ncol >> 7;      // head = col / 128
                    const int w  = ncol & 127;
                    const int bb = mrow >> 11;     // batch = row / SEQ
                    const int ss = mrow & (SEQ - 1);
                    if (w < HDIM)
                        Kout[(((size_t)(bb * NHEADS + hh)) * SEQ + ss) * HDIM + w] = (__bf16)v;
                    else
                        Vtout[(((size_t)(bb * NHEADS + hh)) * HDIM + (w - HDIM)) * SEQ + ss] = (__bf16)v;
                }
            }
        }
    }
}

// ---------------------------------------------------------------------------
// Flash attention over one (b,h): block = 128 threads = 4 waves,
// each wave owns a 16-query strip; iterates keys in 32-wide tiles.
// All K/V fragment loads for a tile are issued up-front so the 16 b128 loads
// clause together and their latency hides under score WMMAs + softmax.
// Q: [B,S,D] bf16; K: [B,H,S,hd] bf16; V^T: [B,H,hd,S] bf16; Ctx: [B,S,D] bf16
// ---------------------------------------------------------------------------
__global__ __launch_bounds__(128)
void attn_wmma_kernel(const __bf16* __restrict__ Q, const __bf16* __restrict__ Kb,
                      const __bf16* __restrict__ Vt, __bf16* __restrict__ Ctx)
{
    __shared__ __align__(16) __bf16 P[4 * 16 * 40];  // per-wave P tile [16][32] pad 40

    const int lane = threadIdx.x & 31;
    const int wv   = threadIdx.x >> 5;
    const int half = lane >> 4;
    const int lm   = lane & 15;
    const int h    = blockIdx.y;
    const int b    = blockIdx.z;
    const int q0   = blockIdx.x * 64 + wv * 16;

    // Q A-fragments for d=0..31 and d=32..63 (register-resident throughout)
    const __bf16* qrow = Q + ((size_t)b * SEQ + q0 + lm) * D_MODEL + h * HDIM;
    const v16bf aq0 = ld_frag(qrow + half * 16, 8);
    const v16bf aq1 = ld_frag(qrow + 32 + half * 16, 8);

    const __bf16* Kh = Kb + (size_t)(b * NHEADS + h) * SEQ * HDIM;
    const __bf16* Vh = Vt + (size_t)(b * NHEADS + h) * HDIM * SEQ;
    __bf16* Pw = &P[wv * 16 * 40];

    float mr[8], lr[8];
#pragma unroll
    for (int j = 0; j < 8; ++j) { mr[j] = -3.0e38f; lr[j] = 0.0f; }
    v8f o0 = zero8(), o1 = zero8(), o2 = zero8(), o3 = zero8();

    for (int kt = 0; kt < SEQ; kt += 32) {
        if (kt + 32 < SEQ) {
            __builtin_prefetch(Kh + (size_t)(kt + 32 + lm) * HDIM, 0, 0);
            __builtin_prefetch(Vh + (size_t)lm * SEQ + kt + 32, 0, 0);
        }

        // ---- issue ALL fragment loads for this tile up front (16 x b128) ----
        const __bf16* k0p = Kh + (size_t)(kt + lm) * HDIM + half * 16;
        const __bf16* k1p = Kh + (size_t)(kt + 16 + lm) * HDIM + half * 16;
        const v16bf bk00 = ld_frag(k0p, 8);            // keys kt..+15,  d 0..31
        const v16bf bk01 = ld_frag(k0p + 32, 8);       // keys kt..+15,  d 32..63
        const v16bf bk10 = ld_frag(k1p, 8);            // keys kt+16..+31, d 0..31
        const v16bf bk11 = ld_frag(k1p + 32, 8);       // keys kt+16..+31, d 32..63
        const __bf16* vp = Vh + (size_t)lm * SEQ + kt + half * 16;
        const v16bf bv0 = ld_frag(vp, 8);                        // d tile 0
        const v16bf bv1 = ld_frag(vp + (size_t)16 * SEQ, 8);     // d tile 1
        const v16bf bv2 = ld_frag(vp + (size_t)32 * SEQ, 8);     // d tile 2
        const v16bf bv3 = ld_frag(vp + (size_t)48 * SEQ, 8);     // d tile 3

        // ---- scores: two 16x16 C fragments ----
        v8f c0 = zero8(), c1 = zero8();
        c0 = wmma_bf16(aq0, bk00, c0);
        c0 = wmma_bf16(aq1, bk01, c0);
        c1 = wmma_bf16(aq0, bk10, c1);
        c1 = wmma_bf16(aq1, bk11, c1);

        // ---- online softmax; row r = half*8+j lives in 16 lanes of this half
        float mx[8], sm[8], cr[8];
#pragma unroll
        for (int j = 0; j < 8; ++j) {
            c0[j] *= 0.125f;  // 1/sqrt(HDIM)
            c1[j] *= 0.125f;
            mx[j] = fmaxf(c0[j], c1[j]);
        }
#pragma unroll
        for (int d = 1; d < 16; d <<= 1)
#pragma unroll
            for (int j = 0; j < 8; ++j) mx[j] = fmaxf(mx[j], __shfl_xor(mx[j], d));
#pragma unroll
        for (int j = 0; j < 8; ++j) {
            const float mn = fmaxf(mr[j], mx[j]);
            cr[j] = __expf(mr[j] - mn);
            mr[j] = mn;
            const float p0 = __expf(c0[j] - mn);
            const float p1 = __expf(c1[j] - mn);
            c0[j] = p0; c1[j] = p1;
            sm[j] = p0 + p1;
        }
#pragma unroll
        for (int d = 1; d < 16; d <<= 1)
#pragma unroll
            for (int j = 0; j < 8; ++j) sm[j] += __shfl_xor(sm[j], d);
#pragma unroll
        for (int j = 0; j < 8; ++j) {
            lr[j] = lr[j] * cr[j] + sm[j];
            o0[j] *= cr[j]; o1[j] *= cr[j]; o2[j] *= cr[j]; o3[j] *= cr[j];
            // C-layout -> LDS [row][key] for re-read as A fragment
            Pw[(half * 8 + j) * 40 + lm]      = (__bf16)c0[j];
            Pw[(half * 8 + j) * 40 + 16 + lm] = (__bf16)c1[j];
        }
        __asm__ volatile("s_wait_dscnt 0x0" ::: "memory");  // wave-local LDS RAW

        const v16bf pa = ld_frag(Pw + lm * 40 + half * 16, 8);
        o0 = wmma_bf16(pa, bv0, o0);
        o1 = wmma_bf16(pa, bv1, o1);
        o2 = wmma_bf16(pa, bv2, o2);
        o3 = wmma_bf16(pa, bv3, o3);
    }

#pragma unroll
    for (int j = 0; j < 8; ++j) lr[j] = 1.0f / lr[j];
    __bf16* cbase = Ctx + ((size_t)b * SEQ + q0) * D_MODEL + h * HDIM;
#pragma unroll
    for (int j = 0; j < 8; ++j) {
        __bf16* crow = cbase + (size_t)(half * 8 + j) * D_MODEL;
        crow[0 * 16 + lm] = (__bf16)(o0[j] * lr[j]);
        crow[1 * 16 + lm] = (__bf16)(o1[j] * lr[j]);
        crow[2 * 16 + lm] = (__bf16)(o2[j] * lr[j]);
        crow[3 * 16 + lm] = (__bf16)(o3[j] * lr[j]);
    }
}

// ---------------------------------------------------------------------------
extern "C" void kernel_launch(void* const* d_in, const int* in_sizes, int n_in,
                              void* d_out, int out_size, void* d_ws, size_t ws_size,
                              hipStream_t stream) {
    (void)in_sizes; (void)n_in; (void)out_size; (void)ws_size;
    const float* x   = (const float*)d_in[0];
    const float* y   = (const float*)d_in[1];
    const float* Wkv = (const float*)d_in[2];
    const float* bkv = (const float*)d_in[3];
    const float* Wq  = (const float*)d_in[4];
    const float* bq  = (const float*)d_in[5];
    const float* Wo  = (const float*)d_in[6];
    const float* bo  = (const float*)d_in[7];
    float* out = (float*)d_out;

    char* ws = (char*)d_ws;
    __bf16* Qb  = (__bf16*)(ws);                       // [B*S, D]      8 MB
    __bf16* Kb  = (__bf16*)(ws + (8u  << 20));         // [B,H,S,hd]    8 MB
    __bf16* Vtb = (__bf16*)(ws + (16u << 20));         // [B,H,hd,S]    8 MB
    __bf16* Ctx = (__bf16*)(ws + (24u << 20));         // [B*S, D]      8 MB

    const int M = BATCH * SEQ;  // 4096

    // kv = x @ Wkv + bkv  -> scatter into K and V^T
    gemm_wmma_kernel<1, false><<<dim3(M / 128, 2 * D_MODEL / 64), 256, 0, stream>>>(
        x, nullptr, Wkv, bkv, nullptr, nullptr, Kb, Vtb, M, 2 * D_MODEL, D_MODEL);
    // q = y @ Wq + bq -> bf16 row-major
    gemm_wmma_kernel<0, false><<<dim3(M / 128, D_MODEL / 64), 256, 0, stream>>>(
        y, nullptr, Wq, bq, Qb, nullptr, nullptr, nullptr, M, D_MODEL, D_MODEL);
    // flash attention -> Ctx
    attn_wmma_kernel<<<dim3(SEQ / 64, NHEADS, BATCH), 128, 0, stream>>>(Qb, Kb, Vtb, Ctx);
    // out = Ctx @ Wo + bo -> f32
    gemm_wmma_kernel<2, true><<<dim3(M / 128, D_MODEL / 64), 256, 0, stream>>>(
        nullptr, Ctx, Wo, bo, nullptr, out, nullptr, nullptr, M, D_MODEL, D_MODEL);
}
